// iPVAE_69578470195593
// MI455X (gfx1250) — compile-verified, matrix operand order checked by reading, not run on hardware
//
#include <hip/hip_runtime.h>
#include <hip/hip_bf16.h>
#include <stdint.h>

#define B_ROWS   65536
#define ZDIM     256
#define XDIM     784
#define XPAD     800            // XDIM padded to multiple of 32 (K for c-GEMM)
#define KB_X     (XPAD/32)      // 25
#define KB_Z     (ZDIM/32)      // 8
#define STEPS    16
#define ALPHA    0.1f

#define WAVES_PER_WG 4
#define ROWS_PER_WG  (WAVES_PER_WG * 16)
#define ZT_STRIDE    264        // padded LDS row stride (halves) to spread banks

// ws layout (bytes): [0, 409600) = swizzled alpha*W bf16 (25*16*32*16 halves)
//                    [409600, 540672) = swizzled alpha*G bf16 (8*16*32*16 halves)
#define GSWZ_OFF 409600

typedef __attribute__((ext_vector_type(16))) __bf16 v16bf;
typedef __attribute__((ext_vector_type(8)))  __bf16 v8bf;
typedef __attribute__((ext_vector_type(8)))  float  v8f;

// ---------- helpers ----------
__device__ __forceinline__ __bf16 to_bf16(float f) {
    return static_cast<__bf16>(f);      // native v_cvt_*_bf16_f32 on gfx1250
}

// B-fragment swizzle index for a (K x 256) bf16 matrix, K multiple of 32.
// Fragment (kb,nt): lane holds 16 contiguous halves at ((kb*16+nt)*32+lane)*16.
__device__ __forceinline__ uint32_t swz_idx(uint32_t k, uint32_t n) {
    uint32_t kb   = k >> 5;
    uint32_t km   = k & 31u;
    uint32_t hi   = (km >> 3) & 1u;                       // lane >= 16 half
    uint32_t e    = (km & 7u) | (((km >> 4) & 1u) << 3);  // element within v16bf
    uint32_t lane = (hi << 4) | (n & 15u);
    uint32_t nt   = n >> 4;
    return (((kb * 16u + nt) * 32u + lane) << 4) + e;
}

__device__ __forceinline__ uint32_t whash(uint32_t x) {
    x ^= x >> 16; x *= 0x7feb352du;
    x ^= x >> 15; x *= 0x846ca68bu;
    x ^= x >> 16; return x;
}
__device__ __forceinline__ float u01(uint32_t h) {
    return (float)(h >> 8) * (1.0f / 16777216.0f) + (0.5f / 16777216.0f);
}

// Deterministic counter-based Poisson sampler (Knuth small-rate, normal approx large-rate)
__device__ __attribute__((noinline)) float poisson_sample(uint32_t seed, float r) {
    if (!(r > 1e-10f)) return 0.0f;
    if (r > 64.0f) {
        float a = u01(whash(seed ^ 0x68bc21ebu));
        float c = u01(whash(seed ^ 0x02e5be93u));
        float g = sqrtf(-2.0f * __logf(a)) * __cosf(6.2831853f * c);
        return fmaxf(0.0f, floorf(r + sqrtf(r) * g + 0.5f));
    }
    float L = __expf(-r);
    uint32_t st = seed;
    float p = 1.0f;
    int k = 0;
    do {
        st = whash(st + 0x9E3779B9u);
        p *= u01(st);
        ++k;
    } while (p > L && k < 512);
    return (float)(k - 1);
}

// ---------- prep kernel 1: swizzle alpha*W into B-fragment order (K padded to 800) ----------
__global__ void prep_swz_W(const float* __restrict__ W, __bf16* __restrict__ Wswz) {
    uint32_t tid = blockIdx.x * blockDim.x + threadIdx.x;     // XPAD*ZDIM threads
    uint32_t k = tid / ZDIM;          // x index (padded)
    uint32_t n = tid % ZDIM;          // z index
    float v = (k < XDIM) ? (ALPHA * W[(size_t)k * ZDIM + n]) : 0.0f;
    Wswz[swz_idx(k, n)] = to_bf16(v);
}

// ---------- prep kernel 2: G = alpha * W^T W, swizzled into B-fragment order ----------
__global__ void prep_swz_G(const float* __restrict__ W, __bf16* __restrict__ Gswz) {
    uint32_t tid = blockIdx.x * blockDim.x + threadIdx.x;     // ZDIM*ZDIM threads
    uint32_t k = tid / ZDIM;          // z1
    uint32_t n = tid % ZDIM;          // z2
    float acc = 0.0f;
    for (int xx = 0; xx < XDIM; ++xx)
        acc += W[(size_t)xx * ZDIM + k] * W[(size_t)xx * ZDIM + n];
    Gswz[swz_idx(k, n)] = to_bf16(ALPHA * acc);
}

// ---------- main fused kernel ----------
__global__ __launch_bounds__(WAVES_PER_WG * 32, 1)
void ipvae_main(const float* __restrict__ x, const float* __restrict__ u0,
                const float* __restrict__ bvec,
                const __bf16* __restrict__ Wswz, const __bf16* __restrict__ Gswz,
                float* __restrict__ out) {
    __shared__ alignas(16) __bf16 zT[WAVES_PER_WG][16][ZT_STRIDE];

    const int wave  = threadIdx.x >> 5;
    const int lane  = threadIdx.x & 31;
    const int laneN = lane & 15;
    const int hi    = lane >> 4;
    const int off8  = hi * 8;
    const int rowBase = blockIdx.x * ROWS_PER_WG + wave * 16;

    // ---- one-time: c[nt] = alpha*(x - b) @ W fragments (C/D layout), via WMMA ----
    v8f c[16];
#pragma unroll
    for (int nt = 0; nt < 16; ++nt) c[nt] = (v8f)(0.0f);

    const float* xrow = x + (size_t)(rowBase + laneN) * XDIM;
    for (int kb = 0; kb < KB_X; ++kb) {
        const int base = kb * 32 + off8;
        const bool pad = (kb == KB_X - 1);      // K >= 784 in second chunk
        float4 z4 = make_float4(0.f, 0.f, 0.f, 0.f);
        float4 x0a = *(const float4*)(xrow + base);
        float4 x0b = *(const float4*)(xrow + base + 4);
        float4 x1a = pad ? z4 : *(const float4*)(xrow + base + 16);
        float4 x1b = pad ? z4 : *(const float4*)(xrow + base + 20);
        float4 b0a = *(const float4*)(bvec + base);
        float4 b0b = *(const float4*)(bvec + base + 4);
        float4 b1a = pad ? z4 : *(const float4*)(bvec + base + 16);
        float4 b1b = pad ? z4 : *(const float4*)(bvec + base + 20);
        v16bf a;
        a[0]  = to_bf16(x0a.x - b0a.x); a[1]  = to_bf16(x0a.y - b0a.y);
        a[2]  = to_bf16(x0a.z - b0a.z); a[3]  = to_bf16(x0a.w - b0a.w);
        a[4]  = to_bf16(x0b.x - b0b.x); a[5]  = to_bf16(x0b.y - b0b.y);
        a[6]  = to_bf16(x0b.z - b0b.z); a[7]  = to_bf16(x0b.w - b0b.w);
        a[8]  = to_bf16(x1a.x - b1a.x); a[9]  = to_bf16(x1a.y - b1a.y);
        a[10] = to_bf16(x1a.z - b1a.z); a[11] = to_bf16(x1a.w - b1a.w);
        a[12] = to_bf16(x1b.x - b1b.x); a[13] = to_bf16(x1b.y - b1b.y);
        a[14] = to_bf16(x1b.z - b1b.z); a[15] = to_bf16(x1b.w - b1b.w);
#pragma unroll
        for (int nt = 0; nt < 16; ++nt) {
            v16bf bw = *(const v16bf*)(Wswz + (((size_t)kb * 16 + nt) * 32 + lane) * 16);
            c[nt] = __builtin_amdgcn_wmma_f32_16x16x32_bf16(
                        false, a, false, bw, (short)0, c[nt], false, false);
        }
    }

    // Warm the swizzled-G region (128 KB, reused 16x) into cache while phase A runs.
#pragma unroll
    for (int kb = 0; kb < KB_Z; ++kb)
        __builtin_prefetch(Gswz + (((size_t)kb * 16) * 32 + lane) * 16 +
                               (size_t)wave * 4096, 0, 1);

    // ---- init u = broadcast(initial_state) ----
    v8f u[16];
#pragma unroll
    for (int nt = 0; nt < 16; ++nt) {
        float s = u0[nt * 16 + laneN];
#pragma unroll
        for (int j = 0; j < 8; ++j) u[nt][j] = s;
    }

    // ---- 16 inference steps ----
    for (int t = 0; t < STEPS; ++t) {
        // phase A: r = exp(u); z ~ Poisson(r); stage z (bf16) in LDS; u <- u - c
#pragma unroll
        for (int nt = 0; nt < 16; ++nt) {
            const int colg = nt * 16 + laneN;
#pragma unroll
            for (int j = 0; j < 8; ++j) {
                const int rloc = j + hi * 8;
                float uu = u[nt][j];
                float r  = fminf(__expf(uu), 1.0e5f);
                uint32_t seed = (uint32_t)(rowBase + rloc) * 0x85EBCA6Bu
                              ^ (uint32_t)colg * 0xC2B2AE35u
                              ^ (uint32_t)(t + 1) * 0x9E3779B9u;
                float z = poisson_sample(seed, r);
                zT[wave][rloc][colg] = to_bf16(z);
                if (t == STEPS - 1)
                    out[(size_t)(rowBase + rloc) * ZDIM + colg] = z;
                u[nt][j] = uu - c[nt][j];
            }
        }
        __syncthreads();

        // phase B: u += z @ (alpha*G)   (16x16x32 bf16 WMMA, A from LDS, B pre-swizzled)
        for (int kb = 0; kb < KB_Z; ++kb) {
            const __bf16* zr = &zT[wave][laneN][kb * 32 + off8];
            v8bf z0 = *(const v8bf*)zr;
            v8bf z1 = *(const v8bf*)(zr + 16);
            v16bf a = __builtin_shufflevector(z0, z1,
                        0, 1, 2, 3, 4, 5, 6, 7, 8, 9, 10, 11, 12, 13, 14, 15);
#pragma unroll
            for (int nt = 0; nt < 16; ++nt) {
                v16bf g = *(const v16bf*)(Gswz + (((size_t)kb * 16 + nt) * 32 + lane) * 16);
                u[nt] = __builtin_amdgcn_wmma_f32_16x16x32_bf16(
                            false, a, false, g, (short)0, u[nt], false, false);
            }
        }
        __syncthreads();
    }
}

extern "C" void kernel_launch(void* const* d_in, const int* in_sizes, int n_in,
                              void* d_out, int out_size, void* d_ws, size_t ws_size,
                              hipStream_t stream) {
    (void)in_sizes; (void)n_in; (void)out_size; (void)ws_size;
    const float* x  = (const float*)d_in[0];   // [B, 784]
    const float* u0 = (const float*)d_in[1];   // [1, 256]
    const float* W  = (const float*)d_in[2];   // [784, 256]
    const float* b  = (const float*)d_in[3];   // [784]
    // d_in[4] = steps (16), compile-time constant here

    __bf16* Wswz = (__bf16*)((char*)d_ws + 0);
    __bf16* Gswz = (__bf16*)((char*)d_ws + GSWZ_OFF);

    prep_swz_W<<<(XPAD * ZDIM) / 256, 256, 0, stream>>>(W, Wswz);
    prep_swz_G<<<(ZDIM * ZDIM) / 256, 256, 0, stream>>>(W, Gswz);
    ipvae_main<<<B_ROWS / ROWS_PER_WG, WAVES_PER_WG * 32, 0, stream>>>(
        x, u0, b, Wswz, Gswz, (float*)d_out);
}
